// PeptideVAE_29205777612934
// MI455X (gfx1250) — matrix operand order repaired and checked by reading, not compiled, not run on hardware
//
#include <hip/hip_runtime.h>

#define B_  2048
#define T_  64
#define E_  256
#define H_  512
#define Z_  128
#define NV_ 30
#define G4_ 2048  // 4*H

typedef __bf16 v16bf __attribute__((ext_vector_type(16)));
typedef float  v8f   __attribute__((ext_vector_type(8)));

union FragAB { uint4 q[2]; v16bf v; };

__device__ __forceinline__ unsigned short f32_to_bf16(float f) {
    union { float f; unsigned int u; } c; c.f = f;
    unsigned int u = c.u;
    unsigned int r = u + 0x7FFFu + ((u >> 16) & 1u);  // round-to-nearest-even
    return (unsigned short)(r >> 16);
}

__device__ __forceinline__ v8f wmma_bf16(v16bf a, v16bf b, v8f c) {
    return __builtin_amdgcn_wmma_f32_16x16x32_bf16(false, a, false, b, (short)0, c, false, false);
}

// A fragment (16x32 bf16): lane<16 holds K {kc..kc+7, kc+16..kc+23} of row (l15);
// lane>=16 holds K {kc+8..kc+15, kc+24..kc+31}. rowptr is per-lane row base.
__device__ __forceinline__ v16bf load_a(const unsigned short* rowptr, int kc, int hi) {
    FragAB fr;
    const unsigned short* p = rowptr + kc + hi * 8;
    fr.q[0] = *(const uint4*)(p);
    fr.q[1] = *(const uint4*)(p + 16);
    return fr.v;
}

// B fragment (32x16 bf16) from W stored [N][K] row-major:
// lane<16: col n0+l15, K {kc..kc+7, kc+16..kc+23} ; lane>=16: col n0+l15, K {kc+8..kc+15, kc+24..kc+31}.
__device__ __forceinline__ v16bf load_b(const unsigned short* W, int n0, int ld,
                                        int kc, int l15, int hi) {
    FragAB fr;
    const unsigned short* p = W + (size_t)(n0 + l15) * ld + kc + hi * 8;
    fr.q[0] = *(const uint4*)(p);
    fr.q[1] = *(const uint4*)(p + 16);
    return fr.v;
}

// C/D tile store: vgpr v, lane<16 -> (M=v,   N=l15); lane>=16 -> (M=v+8, N=l15)
__device__ __forceinline__ void store_acc_gbuf(float* gbuf, const v8f acc[2][4],
                                               int wm, int wg, int l15, int hi) {
#pragma unroll
    for (int mt = 0; mt < 2; ++mt)
#pragma unroll
        for (int nt = 0; nt < 4; ++nt) {
            int r0 = wm * 32 + mt * 16 + hi * 8;
            int cb = wg * 64 + nt * 16 + l15;
#pragma unroll
            for (int v = 0; v < 8; ++v) gbuf[(r0 + v) * 256 + cb] = acc[mt][nt][v];
        }
}

__device__ __forceinline__ void lstm_epilogue(const float* gbuf, const float* __restrict__ bias,
                                              float* __restrict__ c, unsigned short* __restrict__ hnext,
                                              unsigned short* __restrict__ hidden, int dir,
                                              int m0, int j0, int tid) {
    for (int e = tid; e < 64 * 64; e += 256) {
        int m = e >> 6, j = e & 63;
        int gb = m0 + m, gj = j0 + j;
        float gi = gbuf[m * 256 + j]        + bias[0 * H_ + gj];
        float gf = gbuf[m * 256 + 64 + j]   + bias[1 * H_ + gj];
        float gg = gbuf[m * 256 + 128 + j]  + bias[2 * H_ + gj];
        float go = gbuf[m * 256 + 192 + j]  + bias[3 * H_ + gj];
        size_t idx = (size_t)gb * H_ + gj;
        float cold = c[idx];
        float si = 1.0f / (1.0f + __expf(-gi));
        float sf = 1.0f / (1.0f + __expf(-gf));
        float so = 1.0f / (1.0f + __expf(-go));
        float cn = sf * cold + si * tanhf(gg);
        float hn = so * tanhf(cn);
        c[idx] = cn;
        unsigned short hb = f32_to_bf16(hn);
        hnext[idx] = hb;
        if (hidden) hidden[(size_t)gb * (2 * H_) + dir * H_ + gj] = hb;
    }
}

// ---------------- encoder bi-LSTM step (both directions via blockIdx.z) --------------
// g = h_prev @ W_hh^T + emb[x[:,t]] @ W_ih^T + b  (K = 512 + 256), then cell update.
__global__ __launch_bounds__(256)
void bilstm_step_kernel(const int* __restrict__ x, const unsigned short* __restrict__ emb_bf,
                        const unsigned short* __restrict__ Wih_f, const unsigned short* __restrict__ Whh_f,
                        const float* __restrict__ b_f,
                        const unsigned short* __restrict__ Wih_b, const unsigned short* __restrict__ Whh_b,
                        const float* __restrict__ b_b,
                        const unsigned short* __restrict__ hp_f, unsigned short* __restrict__ hn_f,
                        float* __restrict__ c_f,
                        const unsigned short* __restrict__ hp_b, unsigned short* __restrict__ hn_b,
                        float* __restrict__ c_b,
                        unsigned short* __restrict__ hidden, int t) {
    __shared__ float gbuf[64 * 256];
    const int dir = blockIdx.z;
    const int tt  = dir ? (T_ - 1 - t) : t;
    const unsigned short* Wih = dir ? Wih_b : Wih_f;
    const unsigned short* Whh = dir ? Whh_b : Whh_f;
    const unsigned short* hp  = dir ? hp_b : hp_f;
    unsigned short* hn        = dir ? hn_b : hn_f;
    float* c                  = dir ? c_b : c_f;
    const float* bias         = dir ? b_b : b_f;

    const int m0 = blockIdx.x * 64;   // batch tile
    const int j0 = blockIdx.y * 64;   // hidden-unit tile
    const int tid = threadIdx.x;
    const int lane = tid & 31, wid = tid >> 5;
    const int wm = wid >> 2;          // 0..1 (32 batch rows each)
    const int wg = wid & 3;           // gate index i,f,g,o
    const int l15 = lane & 15, hi = lane >> 4;

    v8f acc[2][4];
#pragma unroll
    for (int mt = 0; mt < 2; ++mt)
#pragma unroll
        for (int nt = 0; nt < 4; ++nt) acc[mt][nt] = (v8f)0.0f;

    const unsigned short* arow0 = hp + (size_t)(m0 + wm * 32 + l15) * H_;
    const unsigned short* arow1 = arow0 + (size_t)16 * H_;
    // recurrent part: K = 0..511
#pragma unroll 2
    for (int kc = 0; kc < H_; kc += 32) {
        v16bf a0 = load_a(arow0, kc, hi);
        v16bf a1 = load_a(arow1, kc, hi);
#pragma unroll
        for (int nt = 0; nt < 4; ++nt) {
            v16bf bfr = load_b(Whh, wg * H_ + j0 + nt * 16, H_, kc, l15, hi);
            acc[0][nt] = wmma_bf16(a0, bfr, acc[0][nt]);
            acc[1][nt] = wmma_bf16(a1, bfr, acc[1][nt]);
        }
    }
    // input part: embedding gather, K = 0..255 (emb is L0/L1-resident, 15 KB)
    const int tok0 = x[(m0 + wm * 32 + l15) * T_ + tt];
    const int tok1 = x[(m0 + wm * 32 + 16 + l15) * T_ + tt];
    const unsigned short* erow0 = emb_bf + (size_t)tok0 * E_;
    const unsigned short* erow1 = emb_bf + (size_t)tok1 * E_;
#pragma unroll 2
    for (int kc = 0; kc < E_; kc += 32) {
        v16bf a0 = load_a(erow0, kc, hi);
        v16bf a1 = load_a(erow1, kc, hi);
#pragma unroll
        for (int nt = 0; nt < 4; ++nt) {
            v16bf bfr = load_b(Wih, wg * H_ + j0 + nt * 16, E_, kc, l15, hi);
            acc[0][nt] = wmma_bf16(a0, bfr, acc[0][nt]);
            acc[1][nt] = wmma_bf16(a1, bfr, acc[1][nt]);
        }
    }

    store_acc_gbuf(gbuf, acc, wm, wg, l15, hi);
    __syncthreads();
    lstm_epilogue(gbuf, bias, c, hn, hidden, dir, m0, j0, tid);
}

// ---------------- decoder LSTM step: g = inp @ W_dih^T + h_prev @ W_dhh^T + b_d ------
__global__ __launch_bounds__(256)
void dec_step_kernel(const unsigned short* __restrict__ inp,
                     const unsigned short* __restrict__ hp,
                     const unsigned short* __restrict__ Wdih, const unsigned short* __restrict__ Wdhh,
                     const float* __restrict__ b_d,
                     unsigned short* __restrict__ hn, float* __restrict__ c) {
    __shared__ float gbuf[64 * 256];
    const int m0 = blockIdx.x * 64;
    const int j0 = blockIdx.y * 64;
    const int tid = threadIdx.x;
    const int lane = tid & 31, wid = tid >> 5;
    const int wm = wid >> 2, wg = wid & 3;
    const int l15 = lane & 15, hi = lane >> 4;

    v8f acc[2][4];
#pragma unroll
    for (int mt = 0; mt < 2; ++mt)
#pragma unroll
        for (int nt = 0; nt < 4; ++nt) acc[mt][nt] = (v8f)0.0f;

    const unsigned short* i0 = inp + (size_t)(m0 + wm * 32 + l15) * H_;
    const unsigned short* i1 = i0 + (size_t)16 * H_;
    const unsigned short* h0 = hp + (size_t)(m0 + wm * 32 + l15) * H_;
    const unsigned short* h1 = h0 + (size_t)16 * H_;
#pragma unroll 2
    for (int kc = 0; kc < H_; kc += 32) {
        v16bf a0 = load_a(i0, kc, hi);
        v16bf a1 = load_a(i1, kc, hi);
#pragma unroll
        for (int nt = 0; nt < 4; ++nt) {
            v16bf bfr = load_b(Wdih, wg * H_ + j0 + nt * 16, H_, kc, l15, hi);
            acc[0][nt] = wmma_bf16(a0, bfr, acc[0][nt]);
            acc[1][nt] = wmma_bf16(a1, bfr, acc[1][nt]);
        }
    }
#pragma unroll 2
    for (int kc = 0; kc < H_; kc += 32) {
        v16bf a0 = load_a(h0, kc, hi);
        v16bf a1 = load_a(h1, kc, hi);
#pragma unroll
        for (int nt = 0; nt < 4; ++nt) {
            v16bf bfr = load_b(Wdhh, wg * H_ + j0 + nt * 16, H_, kc, l15, hi);
            acc[0][nt] = wmma_bf16(a0, bfr, acc[0][nt]);
            acc[1][nt] = wmma_bf16(a1, bfr, acc[1][nt]);
        }
    }

    store_acc_gbuf(gbuf, acc, wm, wg, l15, hi);
    __syncthreads();
    lstm_epilogue(gbuf, b_d, c, hn, nullptr, 0, m0, j0, tid);
}

// ---------------- VAE heads: mu/logvar/z from hidden [B,1024] ------------------------
__global__ __launch_bounds__(256)
void heads_kernel(const unsigned short* __restrict__ hidden,
                  const unsigned short* __restrict__ Wmu, const float* __restrict__ bmu,
                  const unsigned short* __restrict__ Wlv, const float* __restrict__ blv,
                  const float* __restrict__ eps,
                  float* __restrict__ out_mu, float* __restrict__ out_lv, float* __restrict__ out_z,
                  unsigned short* __restrict__ z_bf) {
    __shared__ float gbuf[64 * 256];
    const int m0 = blockIdx.x * 64;
    const int tid = threadIdx.x;
    const int lane = tid & 31, wid = tid >> 5;
    const int wm = wid >> 2, wg = wid & 3;   // cols 0..127 = mu, 128..255 = lv
    const int l15 = lane & 15, hi = lane >> 4;

    v8f acc[2][4];
#pragma unroll
    for (int mt = 0; mt < 2; ++mt)
#pragma unroll
        for (int nt = 0; nt < 4; ++nt) acc[mt][nt] = (v8f)0.0f;

    const unsigned short* Wsel[4]; int nsel[4];
#pragma unroll
    for (int nt = 0; nt < 4; ++nt) {
        int ncol = wg * 64 + nt * 16;
        if (ncol < 128) { Wsel[nt] = Wmu; nsel[nt] = ncol; }
        else            { Wsel[nt] = Wlv; nsel[nt] = ncol - 128; }
    }

    const unsigned short* a0p = hidden + (size_t)(m0 + wm * 32 + l15) * (2 * H_);
    const unsigned short* a1p = a0p + (size_t)16 * (2 * H_);
#pragma unroll 2
    for (int kc = 0; kc < 2 * H_; kc += 32) {
        v16bf a0 = load_a(a0p, kc, hi);
        v16bf a1 = load_a(a1p, kc, hi);
#pragma unroll
        for (int nt = 0; nt < 4; ++nt) {
            v16bf bfr = load_b(Wsel[nt], nsel[nt], 2 * H_, kc, l15, hi);
            acc[0][nt] = wmma_bf16(a0, bfr, acc[0][nt]);
            acc[1][nt] = wmma_bf16(a1, bfr, acc[1][nt]);
        }
    }

    store_acc_gbuf(gbuf, acc, wm, wg, l15, hi);
    __syncthreads();
    for (int e = tid; e < 64 * 128; e += 256) {
        int m = e >> 7, j = e & 127;
        int gb = m0 + m;
        float mu = gbuf[m * 256 + j]       + bmu[j];
        float lv = gbuf[m * 256 + 128 + j] + blv[j];
        float z  = mu + eps[(size_t)gb * Z_ + j] * __expf(0.5f * lv);
        size_t idx = (size_t)gb * Z_ + j;
        out_mu[idx] = mu;
        out_lv[idx] = lv;
        out_z[idx]  = z;
        z_bf[idx]   = f32_to_bf16(z);
    }
}

// ---------------- decoder seed: inp0 = z @ W_di^T + b_di  ([B,128]x[128,512]) --------
__global__ __launch_bounds__(256)
void inp0_kernel(const unsigned short* __restrict__ z_bf, const unsigned short* __restrict__ Wdi,
                 const float* __restrict__ b_di, unsigned short* __restrict__ inp0) {
    const int m0 = blockIdx.x * 64;
    const int n0 = blockIdx.y * 128;
    const int tid = threadIdx.x;
    const int lane = tid & 31, wid = tid >> 5;
    const int wm = wid >> 2, wn = wid & 3;
    const int l15 = lane & 15, hi = lane >> 4;

    v8f acc[2][2];
#pragma unroll
    for (int mt = 0; mt < 2; ++mt)
#pragma unroll
        for (int nt = 0; nt < 2; ++nt) acc[mt][nt] = (v8f)0.0f;

    const unsigned short* a0p = z_bf + (size_t)(m0 + wm * 32 + l15) * Z_;
    const unsigned short* a1p = a0p + (size_t)16 * Z_;
#pragma unroll
    for (int kc = 0; kc < Z_; kc += 32) {
        v16bf a0 = load_a(a0p, kc, hi);
        v16bf a1 = load_a(a1p, kc, hi);
#pragma unroll
        for (int nt = 0; nt < 2; ++nt) {
            v16bf bfr = load_b(Wdi, n0 + wn * 32 + nt * 16, Z_, kc, l15, hi);
            acc[0][nt] = wmma_bf16(a0, bfr, acc[0][nt]);
            acc[1][nt] = wmma_bf16(a1, bfr, acc[1][nt]);
        }
    }
#pragma unroll
    for (int mt = 0; mt < 2; ++mt)
#pragma unroll
        for (int nt = 0; nt < 2; ++nt) {
            int col = n0 + wn * 32 + nt * 16 + l15;
#pragma unroll
            for (int v = 0; v < 8; ++v) {
                int row = m0 + wm * 32 + mt * 16 + hi * 8 + v;
                inp0[(size_t)row * H_ + col] = f32_to_bf16(acc[mt][nt][v] + b_di[col]);
            }
        }
}

// ---------------- decoder output projection: out[:,t,:] = h @ W_out^T + b_out --------
__global__ __launch_bounds__(256)
void outproj_kernel(const unsigned short* __restrict__ h, const unsigned short* __restrict__ Wout,
                    const float* __restrict__ b_out, float* __restrict__ out, int t) {
    const int m0 = blockIdx.x * 128;
    const int tid = threadIdx.x;
    const int lane = tid & 31, wid = tid >> 5;
    const int l15 = lane & 15, hi = lane >> 4;

    v8f acc[2];
    acc[0] = (v8f)0.0f; acc[1] = (v8f)0.0f;
    const unsigned short* ap = h + (size_t)(m0 + wid * 16 + l15) * H_;
#pragma unroll 2
    for (int kc = 0; kc < H_; kc += 32) {
        v16bf a = load_a(ap, kc, hi);
#pragma unroll
        for (int nt = 0; nt < 2; ++nt) {
            v16bf bfr = load_b(Wout, nt * 16, H_, kc, l15, hi);
            acc[nt] = wmma_bf16(a, bfr, acc[nt]);
        }
    }
#pragma unroll
    for (int nt = 0; nt < 2; ++nt) {
        int col = nt * 16 + l15;
        if (col < NV_) {
#pragma unroll
            for (int v = 0; v < 8; ++v) {
                int row = m0 + wid * 16 + hi * 8 + v;
                out[((size_t)row * T_ + t) * NV_ + col] = acc[nt][v] + b_out[col];
            }
        }
    }
}

// ---------------- f32 -> bf16 weight conversion --------------------------------------
__global__ void cvt_bf16_kernel(const float* __restrict__ src, unsigned short* __restrict__ dst, int n) {
    for (int i = blockIdx.x * blockDim.x + threadIdx.x; i < n; i += gridDim.x * blockDim.x)
        dst[i] = f32_to_bf16(src[i]);
}

// =====================================================================================
extern "C" void kernel_launch(void* const* d_in, const int* in_sizes, int n_in,
                              void* d_out, int out_size, void* d_ws, size_t ws_size,
                              hipStream_t stream) {
    (void)in_sizes; (void)n_in; (void)out_size; (void)ws_size;
    const int*   x     = (const int*)d_in[0];
    const float* eps   = (const float*)d_in[1];
    const float* emb   = (const float*)d_in[2];
    const float* Wih_f = (const float*)d_in[3];
    const float* Whh_f = (const float*)d_in[4];
    const float* b_f   = (const float*)d_in[5];
    const float* Wih_b = (const float*)d_in[6];
    const float* Whh_b = (const float*)d_in[7];
    const float* b_b   = (const float*)d_in[8];
    const float* Wmu   = (const float*)d_in[9];
    const float* bmu   = (const float*)d_in[10];
    const float* Wlv   = (const float*)d_in[11];
    const float* blv   = (const float*)d_in[12];
    const float* Wdi   = (const float*)d_in[13];
    const float* bdi   = (const float*)d_in[14];
    const float* Wdih  = (const float*)d_in[15];
    const float* Wdhh  = (const float*)d_in[16];
    const float* b_d   = (const float*)d_in[17];
    const float* Wout  = (const float*)d_in[18];
    const float* bout  = (const float*)d_in[19];

    float* out_recon = (float*)d_out;                               // [B,T,V]
    float* out_mu    = out_recon + (size_t)B_ * T_ * NV_;           // [B,Z]
    float* out_lv    = out_mu + (size_t)B_ * Z_;
    float* out_z     = out_lv + (size_t)B_ * Z_;

    // ---- workspace layout (bf16 weights + state) ----
    char* base = (char*)d_ws;
    size_t off = 0;
    auto alloc = [&](size_t bytes) -> char* {
        char* p = base + off;
        off = (off + bytes + 255) & ~(size_t)255;
        return p;
    };
    unsigned short* emb_bf  = (unsigned short*)alloc((size_t)NV_ * E_ * 2);
    unsigned short* Wihf_bf = (unsigned short*)alloc((size_t)G4_ * E_ * 2);
    unsigned short* Whhf_bf = (unsigned short*)alloc((size_t)G4_ * H_ * 2);
    unsigned short* Wihb_bf = (unsigned short*)alloc((size_t)G4_ * E_ * 2);
    unsigned short* Whhb_bf = (unsigned short*)alloc((size_t)G4_ * H_ * 2);
    unsigned short* Wmu_bf  = (unsigned short*)alloc((size_t)Z_ * 2 * H_ * 2);
    unsigned short* Wlv_bf  = (unsigned short*)alloc((size_t)Z_ * 2 * H_ * 2);
    unsigned short* Wdi_bf  = (unsigned short*)alloc((size_t)H_ * Z_ * 2);
    unsigned short* Wdih_bf = (unsigned short*)alloc((size_t)G4_ * H_ * 2);
    unsigned short* Wdhh_bf = (unsigned short*)alloc((size_t)G4_ * H_ * 2);
    unsigned short* Wout_bf = (unsigned short*)alloc((size_t)32 * H_ * 2);  // padded 30->32 rows
    unsigned short* hf[2] = { (unsigned short*)alloc((size_t)B_ * H_ * 2),
                              (unsigned short*)alloc((size_t)B_ * H_ * 2) };
    unsigned short* hb[2] = { (unsigned short*)alloc((size_t)B_ * H_ * 2),
                              (unsigned short*)alloc((size_t)B_ * H_ * 2) };
    unsigned short* hd[2] = { (unsigned short*)alloc((size_t)B_ * H_ * 2),
                              (unsigned short*)alloc((size_t)B_ * H_ * 2) };
    unsigned short* hidden = (unsigned short*)alloc((size_t)B_ * 2 * H_ * 2);
    unsigned short* inp0b  = (unsigned short*)alloc((size_t)B_ * H_ * 2);
    unsigned short* z_bf   = (unsigned short*)alloc((size_t)B_ * Z_ * 2);
    float* c_f = (float*)alloc((size_t)B_ * H_ * 4);
    float* c_b = (float*)alloc((size_t)B_ * H_ * 4);
    float* c_d = (float*)alloc((size_t)B_ * H_ * 4);

    // ---- init state + padded weight rows (capture-safe async memsets) ----
    hipMemsetAsync(Wout_bf, 0, (size_t)32 * H_ * 2, stream);
    hipMemsetAsync(hf[0], 0, (size_t)B_ * H_ * 2, stream);
    hipMemsetAsync(hb[0], 0, (size_t)B_ * H_ * 2, stream);
    hipMemsetAsync(hd[0], 0, (size_t)B_ * H_ * 2, stream);
    hipMemsetAsync(c_f, 0, (size_t)B_ * H_ * 4, stream);
    hipMemsetAsync(c_b, 0, (size_t)B_ * H_ * 4, stream);
    hipMemsetAsync(c_d, 0, (size_t)B_ * H_ * 4, stream);

    // ---- f32 -> bf16 weight conversion ----
    auto cvt = [&](const float* s, unsigned short* d, size_t n) {
        int blocks = (int)((n + 255) / 256); if (blocks > 2048) blocks = 2048;
        cvt_bf16_kernel<<<blocks, 256, 0, stream>>>(s, d, (int)n);
    };
    cvt(emb,  emb_bf,  (size_t)NV_ * E_);
    cvt(Wih_f, Wihf_bf, (size_t)G4_ * E_);
    cvt(Whh_f, Whhf_bf, (size_t)G4_ * H_);
    cvt(Wih_b, Wihb_bf, (size_t)G4_ * E_);
    cvt(Whh_b, Whhb_bf, (size_t)G4_ * H_);
    cvt(Wmu,  Wmu_bf,  (size_t)Z_ * 2 * H_);
    cvt(Wlv,  Wlv_bf,  (size_t)Z_ * 2 * H_);
    cvt(Wdi,  Wdi_bf,  (size_t)H_ * Z_);
    cvt(Wdih, Wdih_bf, (size_t)G4_ * H_);
    cvt(Wdhh, Wdhh_bf, (size_t)G4_ * H_);
    cvt(Wout, Wout_bf, (size_t)NV_ * H_);   // rows 30/31 remain zero

    // ---- encoder: 64 sequential bi-directional LSTM steps ----
    dim3 egrid(B_ / 64, H_ / 64, 2);
    for (int t = 0; t < T_; ++t) {
        bilstm_step_kernel<<<egrid, 256, 0, stream>>>(
            x, emb_bf, Wihf_bf, Whhf_bf, b_f, Wihb_bf, Whhb_bf, b_b,
            hf[t & 1], hf[(t + 1) & 1], c_f,
            hb[t & 1], hb[(t + 1) & 1], c_b,
            hidden, t);
    }

    // ---- VAE heads ----
    heads_kernel<<<dim3(B_ / 64), 256, 0, stream>>>(hidden, Wmu_bf, bmu, Wlv_bf, blv,
                                                    eps, out_mu, out_lv, out_z, z_bf);

    // ---- decoder seed ----
    inp0_kernel<<<dim3(B_ / 64, H_ / 128), 256, 0, stream>>>(z_bf, Wdi_bf, bdi, inp0b);

    // ---- decoder: 64 sequential steps (cell + vocab projection) ----
    dim3 dgrid(B_ / 64, H_ / 64, 1);
    for (int s = 0; s < T_; ++s) {
        const unsigned short* inp = (s == 0) ? inp0b : hd[s & 1];
        dec_step_kernel<<<dgrid, 256, 0, stream>>>(inp, hd[s & 1], Wdih_bf, Wdhh_bf, b_d,
                                                   hd[(s + 1) & 1], c_d);
        outproj_kernel<<<dim3(B_ / 128), 256, 0, stream>>>(hd[(s + 1) & 1], Wout_bf, bout,
                                                           out_recon, s);
    }
}
